// QLinear_88905823027438
// MI455X (gfx1250) — matrix operand order repaired and checked by reading, not compiled, not run on hardware
//
#include <hip/hip_runtime.h>

// ---------------------------------------------------------------------------
// QLinear (quaternion linear layer):  out = X @ H + bias
//   X: (8192, 2048) fp32, W: (512, 2048) fp32, bias: (2048) fp32
//   H[k][n] = sign[k%4][n%4] * W[k/4][comp[k%4][n%4]*512 + n/4]
// Strategy: prebuild H transposed + split into bf16 hi/lo in d_ws, then a
// WMMA bf16 3-pass GEMM (Xh*Hh + Xh*Hl + Xl*Hh) for ~fp32 accuracy.
// B tiles are staged to LDS with gfx1250 async global->LDS copies.
// ---------------------------------------------------------------------------

typedef __attribute__((ext_vector_type(16))) __bf16 v16bf;
typedef __attribute__((ext_vector_type(8)))  float  v8f;
typedef __attribute__((ext_vector_type(4)))  int    v4i;

typedef __attribute__((address_space(1))) v4i gv4i;   // global int4
typedef __attribute__((address_space(3))) v4i lv4i;   // LDS int4

#define N_ROWS 8192
#define IN_F   2048
#define OUT_F  2048

#if __has_builtin(__builtin_amdgcn_global_load_async_to_lds_b128)
#define HAVE_ASYNC_BUILTIN 1
#endif

// --- bf16 helpers -----------------------------------------------------------
__device__ __forceinline__ unsigned short bf_hi(float x) {
    unsigned int u = __float_as_uint(x);
    u += 0x7FFFu + ((u >> 16) & 1u);          // round-to-nearest-even
    return (unsigned short)(u >> 16);
}
__device__ __forceinline__ float bf_to_f(unsigned short h) {
    return __uint_as_float(((unsigned int)h) << 16);
}
__device__ __forceinline__ unsigned int pack2(unsigned short a, unsigned short b) {
    return (unsigned int)a | ((unsigned int)b << 16);
}

// --- async copy helpers ------------------------------------------------------
__device__ __forceinline__ void async_b128_to_lds(const unsigned short* gsrc,
                                                  unsigned short* lds_dst,
                                                  const unsigned short* gbase,
                                                  unsigned int voff_bytes) {
#if defined(HAVE_ASYNC_BUILTIN)
    (void)gbase; (void)voff_bytes;
    __builtin_amdgcn_global_load_async_to_lds_b128(
        (gv4i*)(v4i*)(void*)(uintptr_t)gsrc,
        (lv4i*)(v4i*)(void*)lds_dst,
        0, 0);
#else
    (void)gsrc;
    unsigned int lds_addr = (unsigned int)(uintptr_t)lds_dst;   // low 32 bits = LDS offset
    unsigned long long base = (unsigned long long)(uintptr_t)gbase;
    asm volatile("global_load_async_to_lds_b128 %0, %1, %2"
                 :: "v"(lds_addr), "v"(voff_bytes), "s"(base)
                 : "memory");
#endif
}

__device__ __forceinline__ void wait_asynccnt0() {
#if __has_builtin(__builtin_amdgcn_s_wait_asynccnt)
    __builtin_amdgcn_s_wait_asynccnt(0);
#else
    asm volatile("s_wait_asynccnt 0x0" ::: "memory");
#endif
}

// --- quaternion block tables ------------------------------------------------
__device__ __constant__ int   c_comp[4][4] = {{0,1,2,3},{1,0,3,2},{2,3,0,1},{3,2,1,0}};
__device__ __constant__ float c_sign[4][4] = {{1.f,-1.f,-1.f,-1.f},
                                              {1.f, 1.f,-1.f, 1.f},
                                              {1.f, 1.f, 1.f,-1.f},
                                              {1.f,-1.f, 1.f, 1.f}};

// ---------------------------------------------------------------------------
// Kernel 1: build H^T split into bf16 hi/lo.
//   HhT/HlT are (OUT_F, IN_F) row-major, i.e. HhT[n*IN_F + k] ~ H[k][n].
// ---------------------------------------------------------------------------
__global__ __launch_bounds__(256) void build_ht(
        const float* __restrict__ W,
        unsigned short* __restrict__ HhT,
        unsigned short* __restrict__ HlT) {
    const int t  = blockIdx.x * 256 + threadIdx.x;   // [0, 2048*512)
    const int n  = t >> 9;                           // output column
    const int ri = t & 511;                          // weight row
    const int kb = ri << 2;                          // k base
    const int qj = n & 3;
    const int cj = n >> 2;

    unsigned short h[4], l[4];
#pragma unroll
    for (int e = 0; e < 4; ++e) {
        const float v = c_sign[e][qj] * W[ri * OUT_F + c_comp[e][qj] * 512 + cj];
        h[e] = bf_hi(v);
        l[e] = bf_hi(v - bf_to_f(h[e]));
    }
    uint2 hv, lv;
    hv.x = pack2(h[0], h[1]); hv.y = pack2(h[2], h[3]);
    lv.x = pack2(l[0], l[1]); lv.y = pack2(l[2], l[3]);
    *(uint2*)(HhT + (size_t)n * IN_F + kb) = hv;
    *(uint2*)(HlT + (size_t)n * IN_F + kb) = lv;
}

// ---------------------------------------------------------------------------
// Kernel 2: tiled WMMA GEMM.
//   Block: 256 threads = 8 waves. Output tile 128(M) x 128(N), BK = 32.
//   Wave grid 4(M) x 2(N): each wave owns 32x64 = 2x4 16x16 accumulators.
// ---------------------------------------------------------------------------
#define BKP 40   // padded LDS row length (elements); 80B row => 16B aligned

union Frag { uint4 u[2]; v16bf v; };

__global__ __launch_bounds__(256) void qgemm(
        const float* __restrict__ X,
        const unsigned short* __restrict__ HhT,
        const unsigned short* __restrict__ HlT,
        const float* __restrict__ bias,
        float* __restrict__ out) {
    __shared__ unsigned short Ah[128 * BKP];
    __shared__ unsigned short Al[128 * BKP];
    __shared__ unsigned short Bh[128 * BKP];
    __shared__ unsigned short Bl[128 * BKP];

    const int m0   = blockIdx.y * 128;
    const int n0   = blockIdx.x * 128;
    const int tid  = threadIdx.x;
    const int lane = tid & 31;
    const int wave = tid >> 5;
    const int wm   = (wave & 3) << 5;    // 0,32,64,96
    const int wn   = (wave >> 2) << 6;   // 0,64
    const int half = lane >> 4;          // 0 or 1
    const int lm   = lane & 15;

    // staging maps
    const int ar = tid >> 3;             // A: row 0..31 (+32 strides), 8 thr/row
    const int ac = (tid & 7) << 2;       // A: float4 col offset 0..28
    const int br = tid >> 2;             // B: row 0..63 (+64 stride), 4 thr/row
    const int bc = (tid & 3) << 3;       // B: 8-elem col offset 0..24

    const v8f vzero = {0.f,0.f,0.f,0.f,0.f,0.f,0.f,0.f};
    v8f acc[2][4];
#pragma unroll
    for (int i = 0; i < 2; ++i)
#pragma unroll
        for (int j = 0; j < 4; ++j) acc[i][j] = vzero;

    for (int k0 = 0; k0 < IN_F; k0 += 32) {
        __syncthreads();   // previous iteration's LDS reads done

        // ---- stage B tile via async global->LDS (already bf16, 64B rows) ----
#pragma unroll
        for (int rr = 0; rr < 2; ++rr) {
            const int r = br + rr * 64;
            const unsigned int eoff = (unsigned int)((n0 + r) * IN_F + k0 + bc);
            async_b128_to_lds(HhT + eoff, Bh + r * BKP + bc, HhT, eoff * 2u);
            async_b128_to_lds(HlT + eoff, Bl + r * BKP + bc, HlT, eoff * 2u);
        }

        // ---- stage A tile: fp32 -> bf16 hi/lo (128 x 32) through VGPRs ----
#pragma unroll
        for (int rr = 0; rr < 4; ++rr) {
            const int r = ar + rr * 32;
            const float4 f = *(const float4*)(X + (size_t)(m0 + r) * IN_F + k0 + ac);
            unsigned short h0 = bf_hi(f.x), h1 = bf_hi(f.y), h2 = bf_hi(f.z), h3 = bf_hi(f.w);
            unsigned short l0 = bf_hi(f.x - bf_to_f(h0));
            unsigned short l1 = bf_hi(f.y - bf_to_f(h1));
            unsigned short l2 = bf_hi(f.z - bf_to_f(h2));
            unsigned short l3 = bf_hi(f.w - bf_to_f(h3));
            uint2 hv; hv.x = pack2(h0, h1); hv.y = pack2(h2, h3);
            uint2 lv; lv.x = pack2(l0, l1); lv.y = pack2(l2, l3);
            *(uint2*)(Ah + r * BKP + ac) = hv;
            *(uint2*)(Al + r * BKP + ac) = lv;
        }
        if (k0 + 32 < IN_F) {  // prefetch next X chunk into cache
            __builtin_prefetch(X + (size_t)(m0 + ar) * IN_F + (k0 + 32) + ac, 0, 1);
        }

        wait_asynccnt0();      // async B-tile writes to LDS complete
        __syncthreads();

        // ---- load fragments ----
        // A (16-bit 16x32): lane = M; V0..3 = K(0..7 | 8..15), V4..7 = K(16..23 | 24..31)
        Frag afh[2], afl[2], bfh[4], bfl[4];
#pragma unroll
        for (int sm = 0; sm < 2; ++sm) {
            const int m = wm + sm * 16 + lm;
            afh[sm].u[0] = *(const uint4*)(Ah + m * BKP + half * 8);
            afh[sm].u[1] = *(const uint4*)(Ah + m * BKP + 16 + half * 8);
            afl[sm].u[0] = *(const uint4*)(Al + m * BKP + half * 8);
            afl[sm].u[1] = *(const uint4*)(Al + m * BKP + 16 + half * 8);
        }
        // B (32x16): lane = N; V0..3 = K(0..7 | 8..15), V4..7 = K(16..23 | 24..31)
#pragma unroll
        for (int sn = 0; sn < 4; ++sn) {
            const int n = wn + sn * 16 + lm;
            const int ko = half * 8;
            bfh[sn].u[0] = *(const uint4*)(Bh + n * BKP + ko);
            bfh[sn].u[1] = *(const uint4*)(Bh + n * BKP + ko + 16);
            bfl[sn].u[0] = *(const uint4*)(Bl + n * BKP + ko);
            bfl[sn].u[1] = *(const uint4*)(Bl + n * BKP + ko + 16);
        }

        // ---- 3-pass bf16 WMMA: Xh*Hh + Xh*Hl + Xl*Hh ----
#pragma unroll
        for (int sm = 0; sm < 2; ++sm) {
#pragma unroll
            for (int sn = 0; sn < 4; ++sn) {
                acc[sm][sn] = __builtin_amdgcn_wmma_f32_16x16x32_bf16(
                    false, afh[sm].v, false, bfh[sn].v, (short)0, acc[sm][sn], false, false);
                acc[sm][sn] = __builtin_amdgcn_wmma_f32_16x16x32_bf16(
                    false, afh[sm].v, false, bfl[sn].v, (short)0, acc[sm][sn], false, false);
                acc[sm][sn] = __builtin_amdgcn_wmma_f32_16x16x32_bf16(
                    false, afl[sm].v, false, bfh[sn].v, (short)0, acc[sm][sn], false, false);
            }
        }
    }

    // ---- epilogue: bias add + store fp32 ----
    // C layout: VGPR r, lane l -> M = r + (l/16)*8, N = l%16
#pragma unroll
    for (int sn = 0; sn < 4; ++sn) {
        const int col = n0 + wn + sn * 16 + lm;
        const float bv = bias[col];
#pragma unroll
        for (int sm = 0; sm < 2; ++sm) {
            const int mrow = m0 + wm + sm * 16 + half * 8;
#pragma unroll
            for (int r = 0; r < 8; ++r) {
                out[(size_t)(mrow + r) * OUT_F + col] = acc[sm][sn][r] + bv;
            }
        }
    }
}

// ---------------------------------------------------------------------------
extern "C" void kernel_launch(void* const* d_in, const int* in_sizes, int n_in,
                              void* d_out, int out_size, void* d_ws, size_t ws_size,
                              hipStream_t stream) {
    (void)in_sizes; (void)n_in; (void)out_size; (void)ws_size;
    const float* X    = (const float*)d_in[0];   // (8192, 2048)
    const float* W    = (const float*)d_in[1];   // (512, 2048)
    const float* bias = (const float*)d_in[2];   // (2048)
    float* out = (float*)d_out;

    unsigned short* HhT = (unsigned short*)d_ws;              // 8 MB
    unsigned short* HlT = HhT + (size_t)OUT_F * IN_F;         // 8 MB

    build_ht<<<(OUT_F * (IN_F / 4)) / 256, 256, 0, stream>>>(W, HhT, HlT);

    dim3 grid(OUT_F / 128, N_ROWS / 128);   // (16, 64)
    qgemm<<<grid, 256, 0, stream>>>(X, HhT, HlT, bias, out);
}
